// FVTRStage_32993938767897
// MI455X (gfx1250) — matrix operand order repaired and checked by reading, not compiled, not run on hardware
//
#include <hip/hip_runtime.h>
#include <hip/hip_bf16.h>

// ---------------------------------------------------------------------------
// FVTR stage for MI455X (gfx1250, wave32, WMMA).
// Compute-bound (~250 GFLOP vs ~25MB activations) -> everything through
// v_wmma_f32_16x16x32_f16 with f32 accumulation; LDS macro-tiling for reuse.
// All staging is b128 global traffic with multiple loads in flight.
// ---------------------------------------------------------------------------

typedef _Float16 half_t;
typedef __attribute__((ext_vector_type(16))) _Float16 v16h;
typedef __attribute__((ext_vector_type(8)))  _Float16 v8h;
typedef __attribute__((ext_vector_type(4)))  _Float16 v4h;
typedef __attribute__((ext_vector_type(8)))  float    v8f;

#define E_DIM 192
#define NHEAD 6
#define HD    32
#define B_SZ  64
#define H_SZ  8
#define W_SZ  64
#define SEQ   512            // W*H
#define MROWS (B_SZ * SEQ)   // 32768
#define CO    256
#define MCONV (B_SZ * 4 * W_SZ)  // 16384

// ---- WMMA fragment helpers (contiguous K order, same in A and B) ------------------
// A (16x32 f16): lanes 0-15 hold K 0..15; lanes 16-31 hold K 16..31
__device__ __forceinline__ v16h load_frag_a(const half_t* __restrict__ row, int lane) {
  const int c0 = (lane >> 4) * 16;
  v16h a;
#pragma unroll
  for (int j = 0; j < 16; ++j) a[j] = row[c0 + j];
  return a;
}
// B (32x16 f16): lane%16 selects the N column; 16 contiguous K values,
// lanes 0-15 cover K 0..15, lanes 16-31 cover K 16..31 (caller offsets pointer).
__device__ __forceinline__ v16h load_frag_b(const half_t* __restrict__ p) {
  v16h b;
#pragma unroll
  for (int j = 0; j < 16; ++j) b[j] = p[j];
  return b;
}
__device__ __forceinline__ v8f wmma_f16(v16h a, v16h b, v8f c) {
  return __builtin_amdgcn_wmma_f32_16x16x32_f16(false, a, false, b, (short)0, c,
                                                false, false);
}
__device__ __forceinline__ v4h cvt4(float4 f) {
  v4h h;
  h[0] = (half_t)f.x; h[1] = (half_t)f.y; h[2] = (half_t)f.z; h[3] = (half_t)f.w;
  return h;
}

// ---- layout transform: image [B,E,H,W] -> x [(b*512 + w*8 + h), E] ---------
__global__ __launch_bounds__(256) void k_im2seq(const float* __restrict__ img,
                                                float* __restrict__ x) {
  size_t i = (size_t)blockIdx.x * 256 + threadIdx.x;
  if (i >= (size_t)B_SZ * E_DIM * H_SZ * W_SZ) return;
  int w = i % W_SZ;
  int h = (i / W_SZ) % H_SZ;
  int c = (i / (W_SZ * H_SZ)) % E_DIM;
  int b = i / (W_SZ * H_SZ * E_DIM);
  x[((size_t)b * SEQ + w * H_SZ + h) * E_DIM + c] = img[i];
}

// ---- weight convert: W f32 [N,K] row-major -> Bh f16 [K,N] -----------------
__global__ __launch_bounds__(256) void k_w2h(const float* __restrict__ W,
                                             half_t* __restrict__ Bh, int N, int K) {
  size_t i = (size_t)blockIdx.x * 256 + threadIdx.x;
  if (i >= (size_t)N * K) return;
  int n = i / K, k = i % K;
  Bh[(size_t)k * N + n] = (half_t)W[i];
}

// conv_w [CO, E, 3, 1] -> Bh[(c*3+kh)*CO + co]
__global__ __launch_bounds__(256) void k_convw2h(const float* __restrict__ W,
                                                 half_t* __restrict__ Bh) {
  size_t i = (size_t)blockIdx.x * 256 + threadIdx.x;
  if (i >= (size_t)CO * E_DIM * 3) return;
  int co = i / (E_DIM * 3);
  int rem = i % (E_DIM * 3);
  int c = rem / 3, kh = rem % 3;
  Bh[((size_t)c * 3 + kh) * CO + co] = (half_t)W[i];
}

// ---- in-place LayerNorm: one wave32 per row, E/32 elems per lane -----------
__global__ __launch_bounds__(256) void k_ln(float* __restrict__ x,
                                            const float* __restrict__ g,
                                            const float* __restrict__ bt,
                                            int M, int E) {
  int row = blockIdx.x * 8 + (threadIdx.x >> 5);
  int lane = threadIdx.x & 31;
  if (row >= M) return;
  float* p = x + (size_t)row * E;
  const int per = E / 32;  // 6
  float v[8];
  float mu = 0.f;
#pragma unroll 6
  for (int i = 0; i < per; ++i) { v[i] = p[lane + 32 * i]; mu += v[i]; }
#pragma unroll
  for (int s = 16; s >= 1; s >>= 1) mu += __shfl_xor(mu, s, 32);
  mu /= (float)E;
  float var = 0.f;
#pragma unroll 6
  for (int i = 0; i < per; ++i) { float d = v[i] - mu; var += d * d; }
#pragma unroll
  for (int s = 16; s >= 1; s >>= 1) var += __shfl_xor(var, s, 32);
  float rs = rsqrtf(var / (float)E + 1e-5f);
#pragma unroll 6
  for (int i = 0; i < per; ++i) {
    int c = lane + 32 * i;
    p[c] = (v[i] - mu) * rs * g[c] + bt[c];
  }
}

// ---- WMMA GEMM: C[M,N] = A[M,K](f32->f16) x Bw[K,N](f16) + bias (+epilogue)
// EPI: 0 = bias, 1 = bias+relu, 2 = bias+residual(R)
template <int EPI>
__global__ __launch_bounds__(256) void k_gemm(const float* __restrict__ A,
                                              const half_t* __restrict__ Bw,
                                              const float* __restrict__ bias,
                                              const float* __restrict__ R,
                                              float* __restrict__ C,
                                              int M, int N, int K) {
  __shared__ half_t As[128][48];  // stride 96B -> 32B-aligned fragment rows
  __shared__ half_t Bs[64][48];   // stored transposed: Bs[n][k]
  const int tid = threadIdx.x;
  const int lane = tid & 31, wave = tid >> 5;
  const int wm = wave & 3, wn = wave >> 2;    // 4x2 wave grid over 128x64
  const int m0 = blockIdx.x * 128, n0 = blockIdx.y * 64;

  // A stage decomposition: 1024 float4s, 4 per thread
  const int ar = tid >> 3;            // row pair base (0..31), +32 per i
  const int ak4 = (tid & 7) * 4;      // k offset (0,4,..28)
  // B stage decomposition: 256 v8h chunks, 1 per thread
  const int bk = tid >> 3;            // k row (0..31)
  const int bn8 = (tid & 7) * 8;      // n offset (0,8,..56)

  v8f acc[2][2] = {};

  for (int k0 = 0; k0 < K; k0 += 32) {
    // stage A tile 128x32: 4x b128 loads in flight, then cvt + ds_store_b64
    float4 av[4];
#pragma unroll
    for (int i = 0; i < 4; ++i)
      av[i] = *(const float4*)&A[(size_t)(m0 + ar + 32 * i) * K + k0 + ak4];
    // stage B tile 32x64 transposed: 1x b128 load + 8 b16 scatter stores
    v8h bv = *(const v8h*)&Bw[(size_t)(k0 + bk) * N + n0 + bn8];
#pragma unroll
    for (int i = 0; i < 4; ++i)
      *(v4h*)&As[ar + 32 * i][ak4] = cvt4(av[i]);
#pragma unroll
    for (int j = 0; j < 8; ++j) Bs[bn8 + j][bk] = bv[j];
    __syncthreads();
    if (k0 + 32 < K)  // hint next A tile into cache (global_prefetch_b8)
      __builtin_prefetch(&A[(size_t)(m0 + ar) * K + k0 + 32 + ak4], 0, 1);
#pragma unroll
    for (int i = 0; i < 2; ++i) {
      v16h a = load_frag_a(&As[wm * 32 + i * 16 + (lane & 15)][0], lane);
#pragma unroll
      for (int j = 0; j < 2; ++j) {
        v16h b = load_frag_b(&Bs[wn * 32 + j * 16 + (lane & 15)][(lane >> 4) * 16]);
        acc[i][j] = wmma_f16(a, b, acc[i][j]);
      }
    }
    __syncthreads();
  }
  // epilogue: C/D layout -> VGPR r: lanes 0-15 = M r, lanes 16-31 = M r+8
#pragma unroll
  for (int i = 0; i < 2; ++i)
#pragma unroll
    for (int j = 0; j < 2; ++j)
#pragma unroll
      for (int r = 0; r < 8; ++r) {
        int m = m0 + wm * 32 + i * 16 + ((lane < 16) ? r : r + 8);
        int n = n0 + wn * 32 + j * 16 + (lane & 15);
        float v = acc[i][j][r] + bias[n];
        if (EPI == 1) v = fmaxf(v, 0.f);
        if (EPI == 2) v += R[(size_t)m * N + n];
        C[(size_t)m * N + n] = v;
      }
}

// ---- attention: one block per (batch, head, 32-query chunk) ----------------
// qkv [M, 576] f32 (q|k|v each 192 = 6 heads x 32). out [M,192] f32.
// LOCAL specialization: masked blocks get one branchless v_cndmask/element.
template <int LOCAL>
__global__ __launch_bounds__(256) void k_attn(const float* __restrict__ qkv,
                                              float* __restrict__ out) {
  __shared__ half_t Qs[32][32];    // 2 KB
  __shared__ half_t Sh[32][512];   // 32 KB: scores then softmax probs (f16)
  __shared__ half_t KV[128][32];   // 8 KB: K chunk [key][d]; reused as V^T [d][key]
  half_t(*Vst)[128] = (half_t(*)[128])KV;

  const int tid = threadIdx.x, lane = tid & 31, wave = tid >> 5;
  const int bx = blockIdx.x;
  const int qc = bx & 15;               // query chunk (32 queries)
  const int h = (bx >> 4) % NHEAD;
  const int b = bx / (16 * NHEAD);
  const size_t rowbase = (size_t)b * SEQ;
  const float scale = 0.17677669529663687f;  // 1/sqrt(32)

  // staging decomposition: 8 threads cover one 32-float head row (4 floats each)
  const int sr = tid >> 3;            // row (query/key) 0..31, +32 per i
  const int sd4 = (tid & 7) * 4;      // d offset (0,4,..28)

  // Q chunk 32x32 -> f16 (one b128 load + one ds_store_b64 per thread)
  {
    float4 qv = *(const float4*)&qkv[(rowbase + qc * 32 + sr) * 576 + h * HD + sd4];
    *(v4h*)&Qs[sr][sd4] = cvt4(qv);
  }

  // ---- scores: Q (32x32) x K^T, keys in 4 chunks of 128 ----
  for (int kc = 0; kc < 4; ++kc) {
    __syncthreads();
    {
      float4 kv[4];
#pragma unroll
      for (int i = 0; i < 4; ++i)
        kv[i] = *(const float4*)&qkv[(rowbase + kc * 128 + sr + 32 * i) * 576 + 192 +
                                     h * HD + sd4];
#pragma unroll
      for (int i = 0; i < 4; ++i) *(v4h*)&KV[sr + 32 * i][sd4] = cvt4(kv[i]);
    }
    __syncthreads();
#pragma unroll
    for (int t = 0; t < 2; ++t) {  // 16 tiles (2 x 8), 2 per wave
      int tile = wave * 2 + t;
      int ti = tile >> 3, tj = tile & 7;
      v16h a = load_frag_a(&Qs[ti * 16 + (lane & 15)][0], lane);
      v16h bf = load_frag_b(&KV[tj * 16 + (lane & 15)][(lane >> 4) * 16]);
      v8f acc = {};
      acc = wmma_f16(a, bf, acc);
      const int key_abs = kc * 128 + tj * 16 + (lane & 15);
      const int hk = key_abs & 7, wk = key_abs >> 3;  // s = w*H + h
#pragma unroll
      for (int r = 0; r < 8; ++r) {
        int q = ti * 16 + ((lane < 16) ? r : r + 8);
        float v = acc[r] * scale;
        if (LOCAL) {
          // branchless KH=7 x KW=11 locality window -> v_cndmask
          int sq = qc * 32 + q;
          int hq = sq & 7, wq = sq >> 3;
          int dh = hq - hk; dh = dh < 0 ? -dh : dh;
          int dw = wq - wk; dw = dw < 0 ? -dw : dw;
          bool outside = (dh > 3) | (dw > 5);
          v = outside ? -3.0e38f : v;
        }
        Sh[q][key_abs] = (half_t)v;
      }
    }
  }
  __syncthreads();

  // ---- softmax: 4 rows per wave; each lane owns 16 contiguous halves
  //      (one v16h ds load + one v16h ds store per row per lane) ----
#pragma unroll
  for (int rr = 0; rr < 4; ++rr) {
    int q = wave * 4 + rr;
    v16h hv = *(const v16h*)&Sh[q][lane * 16];
    float vals[16];
    float mx = -3.0e38f;
#pragma unroll
    for (int i = 0; i < 16; ++i) {
      vals[i] = (float)hv[i];
      mx = fmaxf(mx, vals[i]);
    }
#pragma unroll
    for (int s = 16; s >= 1; s >>= 1) mx = fmaxf(mx, __shfl_xor(mx, s, 32));
    float sum = 0.f;
#pragma unroll
    for (int i = 0; i < 16; ++i) {
      vals[i] = __expf(vals[i] - mx);
      sum += vals[i];
    }
#pragma unroll
    for (int s = 16; s >= 1; s >>= 1) sum += __shfl_xor(sum, s, 32);
    float inv = 1.0f / sum;
    v16h pv;
#pragma unroll
    for (int i = 0; i < 16; ++i) pv[i] = (half_t)(vals[i] * inv);
    *(v16h*)&Sh[q][lane * 16] = pv;
  }

  // ---- out = P (32x512) x V (512x32), V chunked 128 keys at a time ----
  v8f oacc = {};
  const int ti = wave & 1, tj = (wave >> 1) & 1;  // waves 0..3 own 4 tiles
  // V staging: 1024 float4s (4/thread); consecutive threads read contiguous
  // bytes of one key row, store transposed into Vst[d][key]
  const int vk = tid >> 3;            // key 0..31, +32 per i
  for (int kt = 0; kt < 4; ++kt) {
    __syncthreads();
    {
      float4 vv[4];
#pragma unroll
      for (int i = 0; i < 4; ++i)
        vv[i] = *(const float4*)&qkv[(rowbase + kt * 128 + vk + 32 * i) * 576 + 384 +
                                     h * HD + sd4];
#pragma unroll
      for (int i = 0; i < 4; ++i) {
        v4h hv = cvt4(vv[i]);
#pragma unroll
        for (int j = 0; j < 4; ++j) Vst[sd4 + j][vk + 32 * i] = hv[j];
      }
    }
    __syncthreads();
    if (wave < 4) {
#pragma unroll
      for (int kb = 0; kb < 4; ++kb) {
        v16h a = load_frag_a(&Sh[ti * 16 + (lane & 15)][kt * 128 + kb * 32], lane);
        v16h bf = load_frag_b(&Vst[tj * 16 + (lane & 15)][kb * 32 + (lane >> 4) * 16]);
        oacc = wmma_f16(a, bf, oacc);
      }
    }
  }
  if (wave < 4) {
#pragma unroll
    for (int r = 0; r < 8; ++r) {
      int q = qc * 32 + ti * 16 + ((lane < 16) ? r : r + 8);
      int d = h * HD + tj * 16 + (lane & 15);
      out[(rowbase + q) * E_DIM + d] = oacc[r];
    }
  }
}

// ---- im2col for merging conv (3,1) stride (2,1) pad (1,0) ------------------
// col[m][c*3+kh], m=(b*4+ho)*64+w, h_in = 2*ho-1+kh
__global__ __launch_bounds__(256) void k_im2col(const float* __restrict__ x,
                                                float* __restrict__ col) {
  size_t i = (size_t)blockIdx.x * 256 + threadIdx.x;
  if (i >= (size_t)MCONV * 576) return;
  int m = i / 576, kidx = i % 576;
  int c = kidx / 3, kh = kidx % 3;
  int w = m & 63, ho = (m >> 6) & 3, b = m >> 8;
  int hin = 2 * ho - 1 + kh;
  float v = 0.f;
  if (hin >= 0 && hin < H_SZ)
    v = x[((size_t)b * SEQ + w * H_SZ + hin) * E_DIM + c];
  col[i] = v;
}

// ---- merging LayerNorm over CO channels, write NCHW output -----------------
__global__ __launch_bounds__(256) void k_merge_ln(const float* __restrict__ y,
                                                  const float* __restrict__ g,
                                                  const float* __restrict__ bt,
                                                  float* __restrict__ out) {
  int row = blockIdx.x * 8 + (threadIdx.x >> 5);  // m in [0, 16384)
  int lane = threadIdx.x & 31;
  if (row >= MCONV) return;
  const float* p = y + (size_t)row * CO;
  float v[8];
  float mu = 0.f;
#pragma unroll
  for (int i = 0; i < 8; ++i) { v[i] = p[lane + 32 * i]; mu += v[i]; }
#pragma unroll
  for (int s = 16; s >= 1; s >>= 1) mu += __shfl_xor(mu, s, 32);
  mu /= (float)CO;
  float var = 0.f;
#pragma unroll
  for (int i = 0; i < 8; ++i) { float d = v[i] - mu; var += d * d; }
#pragma unroll
  for (int s = 16; s >= 1; s >>= 1) var += __shfl_xor(var, s, 32);
  float rs = rsqrtf(var / (float)CO + 1e-5f);
  int b = row >> 8, ho = (row >> 6) & 3, w = row & 63;
#pragma unroll
  for (int i = 0; i < 8; ++i) {
    int co = lane + 32 * i;
    out[(((size_t)b * CO + co) * 4 + ho) * W_SZ + w] = (v[i] - mu) * rs * g[co] + bt[co];
  }
}

// ---------------------------------------------------------------------------
extern "C" void kernel_launch(void* const* d_in, const int* in_sizes, int n_in,
                              void* d_out, int out_size, void* d_ws, size_t ws_size,
                              hipStream_t stream) {
  (void)in_sizes; (void)n_in; (void)out_size; (void)ws_size;
  const float* image  = (const float*)d_in[0];
  const float* in_w   = (const float*)d_in[1];
  const float* in_b   = (const float*)d_in[2];
  const float* out_w  = (const float*)d_in[3];
  const float* out_b  = (const float*)d_in[4];
  const float* ln1_g  = (const float*)d_in[5];
  const float* ln1_b  = (const float*)d_in[6];
  const float* ln2_g  = (const float*)d_in[7];
  const float* ln2_b  = (const float*)d_in[8];
  const float* w1     = (const float*)d_in[9];
  const float* b1     = (const float*)d_in[10];
  const float* w2     = (const float*)d_in[11];
  const float* b2     = (const float*)d_in[12];
  const float* conv_w = (const float*)d_in[13];
  const float* conv_b = (const float*)d_in[14];
  const float* mln_g  = (const float*)d_in[15];
  const float* mln_b  = (const float*)d_in[16];
  float* outp = (float*)d_out;

  // ---- workspace carve-out ----
  char* ws = (char*)d_ws;
  size_t off = 0;
  auto carve = [&](size_t bytes) {
    size_t o = off;
    off += (bytes + 255) & ~(size_t)255;
    return o;
  };
  float*  xf   = (float*)(ws + carve((size_t)MROWS * E_DIM * 4));   // activations
  float*  hbuf = (float*)(ws + carve((size_t)MROWS * 768 * 4));     // qkv / mlp hidden / im2col
  float*  tbuf = (float*)(ws + carve((size_t)MROWS * E_DIM * 4));   // attn out / conv out
  half_t* wq_h = (half_t*)(ws + carve((size_t)6 * 576 * 192 * 2));
  half_t* wo_h = (half_t*)(ws + carve((size_t)6 * 192 * 192 * 2));
  half_t* w1_h = (half_t*)(ws + carve((size_t)6 * 768 * 192 * 2));
  half_t* w2_h = (half_t*)(ws + carve((size_t)6 * 192 * 768 * 2));
  half_t* wc_h = (half_t*)(ws + carve((size_t)576 * 256 * 2));

  auto nb = [](size_t n) { return (unsigned)((n + 255) / 256); };

  // ---- weight conversion f32 [N,K] -> f16 [K,N] ----
  for (int i = 0; i < 6; ++i) {
    k_w2h<<<nb(576 * 192), 256, 0, stream>>>(in_w + (size_t)i * 576 * 192,
                                             wq_h + (size_t)i * 576 * 192, 576, 192);
    k_w2h<<<nb(192 * 192), 256, 0, stream>>>(out_w + (size_t)i * 192 * 192,
                                             wo_h + (size_t)i * 192 * 192, 192, 192);
    k_w2h<<<nb(768 * 192), 256, 0, stream>>>(w1 + (size_t)i * 768 * 192,
                                             w1_h + (size_t)i * 768 * 192, 768, 192);
    k_w2h<<<nb(192 * 768), 256, 0, stream>>>(w2 + (size_t)i * 192 * 768,
                                             w2_h + (size_t)i * 192 * 768, 192, 768);
  }
  k_convw2h<<<nb((size_t)CO * E_DIM * 3), 256, 0, stream>>>(conv_w, wc_h);

  // ---- [B,C,H,W] -> [B*S, E] ----
  k_im2seq<<<nb((size_t)B_SZ * E_DIM * H_SZ * W_SZ), 256, 0, stream>>>(image, xf);

  // ---- 6 mixer blocks ----
  for (int i = 0; i < 6; ++i) {
    const int local = (i >= 2) ? 1 : 0;  // LOCAL = (F,F,T,T,T,T)
    // x = LN1(x)   (in place; residual in reference is post-LN)
    k_ln<<<MROWS / 8, 256, 0, stream>>>(xf, ln1_g + i * 192, ln1_b + i * 192,
                                        MROWS, E_DIM);
    // qkv = x @ in_w^T + in_b
    k_gemm<0><<<dim3(MROWS / 128, 576 / 64), 256, 0, stream>>>(
        xf, wq_h + (size_t)i * 576 * 192, in_b + i * 576, nullptr, hbuf,
        MROWS, 576, 192);
    // attention (masked for local blocks)
    if (local)
      k_attn<1><<<B_SZ * NHEAD * 16, 256, 0, stream>>>(hbuf, tbuf);
    else
      k_attn<0><<<B_SZ * NHEAD * 16, 256, 0, stream>>>(hbuf, tbuf);
    // x = attn @ out_w^T + out_b + x
    k_gemm<2><<<dim3(MROWS / 128, 192 / 64), 256, 0, stream>>>(
        tbuf, wo_h + (size_t)i * 192 * 192, out_b + i * 192, xf, xf,
        MROWS, 192, 192);
    // x = LN2(x)
    k_ln<<<MROWS / 8, 256, 0, stream>>>(xf, ln2_g + i * 192, ln2_b + i * 192,
                                        MROWS, E_DIM);
    // h = relu(x @ w1^T + b1)
    k_gemm<1><<<dim3(MROWS / 128, 768 / 64), 256, 0, stream>>>(
        xf, w1_h + (size_t)i * 768 * 192, b1 + i * 768, nullptr, hbuf,
        MROWS, 768, 192);
    // x = h @ w2^T + b2 + x
    k_gemm<2><<<dim3(MROWS / 128, 192 / 64), 256, 0, stream>>>(
        hbuf, w2_h + (size_t)i * 192 * 768, b2 + i * 192, xf, xf,
        MROWS, 192, 768);
  }

  // ---- merging conv as im2col GEMM + channel LayerNorm -> NCHW out ----
  k_im2col<<<nb((size_t)MCONV * 576), 256, 0, stream>>>(xf, hbuf);
  k_gemm<0><<<dim3(MCONV / 128, CO / 64), 256, 0, stream>>>(
      hbuf, wc_h, conv_b, nullptr, tbuf, MCONV, CO, 576);
  k_merge_ln<<<MCONV / 8, 256, 0, stream>>>(tbuf, mln_g, mln_b, outp);
}
